// GPT2Attention_22969485099106
// MI455X (gfx1250) — matrix operand order repaired and checked by reading, not compiled, not run on hardware
//
#include <hip/hip_runtime.h>

// ---------------------------------------------------------------------------
// GPT2Attention (layer_idx=1 path): q = x@Wq.T, k = v = x (K,V are identity)
// out0: attn_out     [B,T,E]     (f32)
// out1: attn_weights [B,H,T,T]   (f32)
// Strategy: bf16 WMMA (v_wmma_f32_16x16x32_bf16) for all three matmuls,
// fused logits+softmax+PV kernel keeping the 16xT score strip in LDS.
// V is pre-transposed per head (V^T[bh][d][t]) so every WMMA fragment load
// is a contiguous 16B/32B global_load_b128 — no scalar gathers.
// ---------------------------------------------------------------------------

typedef __bf16 bf16;
typedef __attribute__((ext_vector_type(16))) __bf16 v16bf;
typedef __attribute__((ext_vector_type(8)))  __bf16 v8bf;
typedef __attribute__((ext_vector_type(8)))  float  v8f;

#define Bn 2
#define Tn 2048
#define En 1024
#define Hn 16
#define Dn 64
#define M_TOT (Bn * Tn)   // 4096 rows of x (flattened)
#define SROW 2052         // LDS row stride (floats); 2052%64==4 -> bank-safe

union V16U { v16bf v; struct { v8bf lo, hi; } p; };

// ---------------------------------------------------------------------------
// Kernel 1: f32 -> bf16 conversion of x and Wq into workspace
// ---------------------------------------------------------------------------
__global__ void __launch_bounds__(256)
cvt_bf16_kernel(const float* __restrict__ x, const float* __restrict__ wq,
                bf16* __restrict__ xb, bf16* __restrict__ wqb) {
  int i = blockIdx.x * blockDim.x + threadIdx.x;
  if (i < M_TOT * En) xb[i]  = (bf16)x[i];
  if (i < En * En)    wqb[i] = (bf16)wq[i];
}

// ---------------------------------------------------------------------------
// Kernel 1b: per-head transpose of V (= x) into V^T[bh][d][t], bf16.
// Reads coalesced (64 consecutive f32 per 64-thread quarter-block).
// ---------------------------------------------------------------------------
__global__ void __launch_bounds__(256)
vtrans_kernel(const float* __restrict__ x, bf16* __restrict__ xt) {
  const int tchunk = blockIdx.x & 127;      // 16 t-rows per block
  const int bh = blockIdx.x >> 7;           // 0..31
  const int b  = bh >> 4, h = bh & 15;
  const int d  = threadIdx.x & 63;
  const int dt = threadIdx.x >> 6;          // 0..3
  for (int r = 0; r < 4; ++r) {
    const int t = tchunk * 16 + r * 4 + dt;
    const float v = x[((size_t)b * Tn + t) * En + h * Dn + d];
    xt[((size_t)bh * Dn + d) * Tn + t] = (bf16)v;
  }
}

// ---------------------------------------------------------------------------
// Kernel 2: Q = X @ Wq^T  (M=4096, N=1024, K=1024), bf16 in, bf16 out.
// One wave per 16x16 tile. A-frag: lane m=l%16, halves k per ISA 16-bit A
// layout; B-frag: lane n=l%16, same k order as A (two 16B loads per lane).
// ---------------------------------------------------------------------------
__global__ void __launch_bounds__(256)
qproj_kernel(const bf16* __restrict__ xb, const bf16* __restrict__ wqb,
             bf16* __restrict__ qb) {
  const int lane = threadIdx.x & 31;
  const int wave = threadIdx.x >> 5;
  const int l16  = lane & 15;
  const int hi   = lane >> 4;

  const int tile = blockIdx.x * 8 + wave;   // 256*64 tiles total
  const int mt = tile >> 6;                 // 64 N-tiles
  const int nt = tile & 63;
  const int m0 = mt * 16, n0 = nt * 16;

  const bf16* arow = xb  + (size_t)(m0 + l16) * En;
  const bf16* brow = wqb + (size_t)(n0 + l16) * En;

  v8f acc = {};
  for (int k0 = 0; k0 < En; k0 += 32) {
    V16U a, b;
    a.p.lo = *(const v8bf*)(arow + k0 + 8 * hi);
    a.p.hi = *(const v8bf*)(arow + k0 + 16 + 8 * hi);
    b.p.lo = *(const v8bf*)(brow + k0 + 8 * hi);
    b.p.hi = *(const v8bf*)(brow + k0 + 16 + 8 * hi);
    acc = __builtin_amdgcn_wmma_f32_16x16x32_bf16(false, a.v, false, b.v,
                                                  (short)0, acc, false, false);
  }
  // D layout: VGPR r -> row m0 + r + 8*hi, col n0 + l16
  for (int r = 0; r < 8; ++r)
    qb[(size_t)(m0 + r + 8 * hi) * En + n0 + l16] = (bf16)acc[r];
}

// ---------------------------------------------------------------------------
// Kernel 3: fused  S = Q K^T -> softmax -> P (global) -> O = P V (global).
// One workgroup (8 waves) handles 16 query rows x full T=2048 for one (b,h).
// ---------------------------------------------------------------------------
__global__ void __launch_bounds__(256)
attn_kernel(const bf16* __restrict__ xb, const bf16* __restrict__ qb,
            const bf16* __restrict__ xt,
            float* __restrict__ attn_out, float* __restrict__ attn_w) {
  extern __shared__ char smem[];
  float* S      = (float*)smem;            // [16][SROW]   exp/logits strip
  float* opart  = S + 16 * SROW;           // [2][16][64]  split-K partials
  float* red    = opart + 2 * 16 * 64;     // [16][16]     reduction scratch
  float* rowmx  = red + 256;               // [16]
  float* rowinv = rowmx + 16;              // [16]

  const int tid  = threadIdx.x;
  const int lane = tid & 31, wave = tid >> 5;
  const int l16  = lane & 15, hi = lane >> 4;

  const int qt = blockIdx.x & 127;         // 128 row tiles
  const int bh = blockIdx.x >> 7;
  const int b  = bh >> 4, h = bh & 15;
  const int q0 = qt * 16;

  const bf16* xbase = xb + (size_t)b * Tn * En + h * Dn; // k head slice
  const bf16* qbase = qb + (size_t)b * Tn * En + h * Dn;
  const bf16* vtb   = xt + (size_t)bh * Dn * Tn;         // V^T[d][t]

  // --- A fragments (Q rows q0..q0+15, full D=64) loaded once per wave ---
  V16U aq[2];
  {
    const bf16* qrow = qbase + (size_t)(q0 + l16) * En;
    for (int kk = 0; kk < 2; ++kk) {
      aq[kk].p.lo = *(const v8bf*)(qrow + 32 * kk + 8 * hi);
      aq[kk].p.hi = *(const v8bf*)(qrow + 32 * kk + 16 + 8 * hi);
    }
  }

  // --- S = Q K^T across all 128 column tiles (8 waves round-robin) ---
  for (int ct = wave; ct < Tn / 16; ct += 8) {
    const int t0 = ct * 16;
    const bf16* krow = xbase + (size_t)(t0 + l16) * En;
    // prefetch the next tile's K rows for this wave (8 tiles ahead)
    if (ct + 8 < Tn / 16)
      __builtin_prefetch(krow + (size_t)128 * En, 0, 1);
    v8f acc = {};
    for (int kk = 0; kk < 2; ++kk) {
      V16U bf;
      bf.p.lo = *(const v8bf*)(krow + 32 * kk + 8 * hi);
      bf.p.hi = *(const v8bf*)(krow + 32 * kk + 16 + 8 * hi);
      acc = __builtin_amdgcn_wmma_f32_16x16x32_bf16(false, aq[kk].v, false,
                                                    bf.v, (short)0, acc,
                                                    false, false);
    }
    for (int r = 0; r < 8; ++r)
      S[(r + 8 * hi) * SROW + t0 + l16] = acc[r];
  }
  __syncthreads();

  // --- row max ---
  const int rr = tid & 15, jj = tid >> 4;   // 16 threads per row
  {
    float m = -3.4e38f;
    const float* p = S + rr * SROW + jj * 128;
    for (int c = 0; c < 128; ++c) m = fmaxf(m, p[c]);
    red[rr * 16 + jj] = m;
  }
  __syncthreads();
  if (tid < 16) {
    float m = red[tid * 16];
    for (int j = 1; j < 16; ++j) m = fmaxf(m, red[tid * 16 + j]);
    rowmx[tid] = m;
  }
  __syncthreads();

  // --- exp (overwrite S) + row sum ---
  {
    const float mx = rowmx[rr];
    float s = 0.f;
    float* p = S + rr * SROW + jj * 128;
    for (int c = 0; c < 128; ++c) {
      float e = __expf(p[c] - mx);
      p[c] = e;
      s += e;
    }
    red[rr * 16 + jj] = s;
  }
  __syncthreads();
  if (tid < 16) {
    float s = 0.f;
    for (int j = 0; j < 16; ++j) s += red[tid * 16 + j];
    rowinv[tid] = 1.0f / s;
  }
  __syncthreads();

  // --- write normalized attn_weights (float4 vectorized) ---
  {
    float* aw = attn_w + (((size_t)(b * Hn + h) * Tn) + q0) * Tn;
    for (int r = 0; r < 16; ++r) {
      const float inv = rowinv[r];
      const float* sp = S + r * SROW;
      float4* dst = (float4*)(aw + (size_t)r * Tn);
      for (int c4 = tid; c4 < Tn / 4; c4 += 256) {
        float4 v = *(const float4*)(sp + c4 * 4);
        v.x *= inv; v.y *= inv; v.z *= inv; v.w *= inv;
        dst[c4] = v;
      }
    }
  }

  // --- O = exp(S) @ V  (split-K over 2 halves x 4 col tiles of 16) ---
  {
    const int nt = wave & 3, ks = wave >> 2;
    const int d0 = nt * 16;
    const bf16* vrow = vtb + (size_t)(d0 + l16) * Tn;  // V^T row for lane's n
    v8f o = {};
    for (int k0 = ks * 1024; k0 < ks * 1024 + 1024; k0 += 32) {
      // A-frag: exp values from LDS, f32 -> bf16
      V16U a;
      const float* sp = S + l16 * SROW + k0;
      v8f lo8 = *(const v8f*)(sp + 8 * hi);
      v8f hi8 = *(const v8f*)(sp + 16 + 8 * hi);
      for (int i = 0; i < 8; ++i) {
        a.v[i]     = (bf16)lo8[i];
        a.v[i + 8] = (bf16)hi8[i];
      }
      // B-frag: V^T row (contiguous t) -> two 16B loads per lane
      V16U bf;
      bf.p.lo = *(const v8bf*)(vrow + k0 + 8 * hi);
      bf.p.hi = *(const v8bf*)(vrow + k0 + 16 + 8 * hi);
      o = __builtin_amdgcn_wmma_f32_16x16x32_bf16(false, a.v, false, bf.v,
                                                  (short)0, o, false, false);
    }
    for (int r = 0; r < 8; ++r)
      opart[ks * 1024 + (r + 8 * hi) * 64 + d0 + l16] = o[r];
  }
  __syncthreads();
  {
    float* ob = attn_out + ((size_t)b * Tn + q0) * En + h * Dn;
    for (int idx = tid; idx < 1024; idx += 256) {
      const int m = idx >> 6, d = idx & 63;
      const float v = (opart[idx] + opart[1024 + idx]) * rowinv[m];
      ob[(size_t)m * En + d] = v;
    }
  }
}

// ---------------------------------------------------------------------------
extern "C" void kernel_launch(void* const* d_in, const int* in_sizes, int n_in,
                              void* d_out, int out_size, void* d_ws,
                              size_t ws_size, hipStream_t stream) {
  (void)in_sizes; (void)n_in; (void)out_size; (void)ws_size;

  const float* x  = (const float*)d_in[0];   // [B,T,E] f32
  const float* wq = (const float*)d_in[1];   // [E,E]   f32
  // d_in[2]/d_in[3] are identity K/V buffers -> k = v = x by construction.

  bf16* xb  = (bf16*)d_ws;                        // 4096*1024 bf16 (8 MB)
  bf16* wqb = xb + (size_t)M_TOT * En;            // 1024*1024 bf16 (2 MB)
  bf16* qbw = wqb + (size_t)En * En;              // 4096*1024 bf16 (8 MB)
  bf16* xt  = qbw + (size_t)M_TOT * En;           // V^T per head     (8 MB)

  float* attn_out = (float*)d_out;                          // B*T*E
  float* attn_w   = attn_out + (size_t)Bn * Tn * En;        // B*H*T*T

  // 1) bf16 conversion of x and Wq
  cvt_bf16_kernel<<<(M_TOT * En + 255) / 256, 256, 0, stream>>>(x, wq, xb, wqb);

  // 1b) per-head transpose of V
  vtrans_kernel<<<Bn * Hn * (Tn / 16), 256, 0, stream>>>(x, xt);

  // 2) Q = X @ Wq^T   (16384 tiles / 8 waves per block)
  qproj_kernel<<<(M_TOT / 16) * (En / 16) / 8, 256, 0, stream>>>(xb, wqb, qbw);

  // 3) fused logits + softmax + P writeback + PV
  const size_t lds_bytes =
      (size_t)(16 * SROW + 2 * 16 * 64 + 256 + 16 + 16) * sizeof(float);
  attn_kernel<<<Bn * Hn * (Tn / 16), 256, lds_bytes, stream>>>(
      xb, qbw, xt, attn_out, attn_w);
}
